// TripletCenterLoss_15298673508622
// MI455X (gfx1250) — matrix operand order repaired and checked by reading, not compiled, not run on hardware
//
#include <hip/hip_runtime.h>

#define BB 8192
#define DD 512
#define CC 512
#define MARGINF 1.0f

typedef __attribute__((ext_vector_type(16))) __bf16 v16bf;
typedef __attribute__((ext_vector_type(8)))  __bf16 v8bf;
typedef __attribute__((ext_vector_type(8)))  float  v8f;

// ---------------- fp32 -> bf16 hi/lo split ----------------
__global__ void k_split_bf16(const float* __restrict__ x, __bf16* __restrict__ hi,
                             __bf16* __restrict__ lo, int n) {
  int i = blockIdx.x * blockDim.x + threadIdx.x;
  if (i < n) {
    float v = x[i];
    __bf16 h = (__bf16)v;               // RNE
    hi[i] = h;
    lo[i] = (__bf16)(v - (float)h);     // residual
  }
}

// ---------------- exact fp32 squared row norms (one wave per row) ----------------
__global__ void k_row_norm2(const float* __restrict__ x, float* __restrict__ out, int rows) {
  int wave = threadIdx.x >> 5;
  int lane = threadIdx.x & 31;
  int row = blockIdx.x * (blockDim.x >> 5) + wave;
  if (row >= rows) return;
  const float* p = x + (size_t)row * DD;
  float s = 0.f;
  for (int k = lane; k < DD; k += 32) { float v = p[k]; s = fmaf(v, v, s); }
  #pragma unroll
  for (int m = 16; m >= 1; m >>= 1) s += __shfl_xor(s, m, 32);
  if (lane == 0) out[row] = s;
}

// ---------------- class presence histogram ----------------
__global__ void k_zero_counts(int* cnt) { if ((int)threadIdx.x < CC) cnt[threadIdx.x] = 0; }

__global__ void k_count(const int* __restrict__ t, int* __restrict__ cnt) {
  int i = blockIdx.x * blockDim.x + threadIdx.x;
  if (i < BB) atomicAdd(&cnt[t[i]], 1);
}

// ---------------- fused WMMA GEMM + distance + masked hardest pos/neg ----------------
// 4 waves / block, each wave owns a 16-row tile of `inputs` and sweeps all 512 centers.
// Three independent accumulator chains (hh / hl / lh) keep the matrix pipe fed instead
// of serializing 48 WMMAs through one C register block.
__global__ __launch_bounds__(128)
void k_dist_minmax(const __bf16* __restrict__ in_hi, const __bf16* __restrict__ in_lo,
                   const __bf16* __restrict__ ce_hi, const __bf16* __restrict__ ce_lo,
                   const float* __restrict__ nin, const float* __restrict__ nc,
                   const int* __restrict__ cnt, const int* __restrict__ targets,
                   float* __restrict__ ap, float* __restrict__ an) {
  const int wave = threadIdx.x >> 5;
  const int lane = threadIdx.x & 31;
  const int half = lane >> 4;      // which 16-lane half
  const int ln   = lane & 15;
  const int m0   = (blockIdx.x * 4 + wave) * 16;

  // A-fragment rows: lane holds row m0 + (lane&15), K-halves split by `half`
  const __bf16* arow_h = in_hi + (size_t)(m0 + ln) * DD;
  const __bf16* arow_l = in_lo + (size_t)(m0 + ln) * DD;

  // C/D layout: element r of this lane -> output row m0 + 8*half + r, col = ln (in tile)
  float ninr[8]; int tr[8];
  #pragma unroll
  for (int r = 0; r < 8; ++r) {
    int row = m0 + half * 8 + r;
    ninr[r] = nin[row];
    tr[r]   = targets[row];
  }

  float amin[8], aap[8];
  #pragma unroll
  for (int r = 0; r < 8; ++r) { amin[r] = INFINITY; aap[r] = -INFINITY; }

  for (int nt = 0; nt < CC / 16; ++nt) {
    const int n = nt * 16 + ln;                    // this lane's center/column
    const __bf16* brow_h = ce_hi + (size_t)n * DD;
    const __bf16* brow_l = ce_lo + (size_t)n * DD;

    v8f acc_hh = {};
    v8f acc_hl = {};
    v8f acc_lh = {};
    for (int kc = 0; kc < DD / 32; ++kc) {
      const int ka = kc * 32 + half * 16;          // A: two 8-elem chunks (K, K+8)
      const int kb = kc * 32 + half * 16;          // B: one contiguous 16-elem chunk
      v8bf a0h = *(const v8bf*)(arow_h + ka);
      v8bf a1h = *(const v8bf*)(arow_h + ka + 8);
      v8bf a0l = *(const v8bf*)(arow_l + ka);
      v8bf a1l = *(const v8bf*)(arow_l + ka + 8);
      v16bf ah, al;
      #pragma unroll
      for (int e = 0; e < 8; ++e) {
        ah[e] = a0h[e]; ah[8 + e] = a1h[e];
        al[e] = a0l[e]; al[8 + e] = a1l[e];
      }
      v16bf bh = *(const v16bf*)(brow_h + kb);
      v16bf bl = *(const v16bf*)(brow_l + kb);
      // split GEMM: hh + hl + lh (drop ll, ~2^-16 relative); 3 independent chains
      acc_hh = __builtin_amdgcn_wmma_f32_16x16x32_bf16(false, ah, false, bh, (short)0, acc_hh, false, false);
      acc_hl = __builtin_amdgcn_wmma_f32_16x16x32_bf16(false, ah, false, bl, (short)0, acc_hl, false, false);
      acc_lh = __builtin_amdgcn_wmma_f32_16x16x32_bf16(false, al, false, bh, (short)0, acc_lh, false, false);
    }

    const bool present = cnt[n] > 0;
    const float ncn = nc[n];
    #pragma unroll
    for (int r = 0; r < 8; ++r) {
      float s  = acc_hh[r] + (acc_hl[r] + acc_lh[r]);
      float sq = ninr[r] + ncn - 2.0f * s;
      float d  = sqrtf(fmaxf(sq, 1e-12f));
      if (n == tr[r])      aap[r]  = fmaxf(aap[r], d);    // the (unique) positive center
      else if (present)    amin[r] = fminf(amin[r], d);   // hardest negative candidate
    }
  }

  // reduce across the 16 lanes of each half (xor of bits 0..3 stays within the half)
  #pragma unroll
  for (int m = 8; m >= 1; m >>= 1) {
    #pragma unroll
    for (int r = 0; r < 8; ++r) {
      amin[r] = fminf(amin[r], __shfl_xor(amin[r], m, 32));
      aap[r]  = fmaxf(aap[r],  __shfl_xor(aap[r],  m, 32));
    }
  }
  if (ln == 0) {
    #pragma unroll
    for (int r = 0; r < 8; ++r) {
      int row = m0 + half * 8 + r;
      an[row] = amin[r];
      ap[row] = aap[r];
    }
  }
}

// ---------------- final loss / precision reduction ----------------
__global__ void k_finalize(const float* __restrict__ ap, const float* __restrict__ an,
                           float* __restrict__ out) {
  __shared__ float sl[256], sp[256];
  float L = 0.f, P = 0.f;
  for (int i = threadIdx.x; i < BB; i += 256) {
    float a = ap[i], m = an[i];
    L += fmaxf(0.f, a - m + MARGINF);   // a - inf + 1 = -inf -> 0, matches reference
    P += (m > a) ? 1.f : 0.f;
  }
  sl[threadIdx.x] = L; sp[threadIdx.x] = P;
  __syncthreads();
  for (int s = 128; s >= 1; s >>= 1) {
    if ((int)threadIdx.x < s) {
      sl[threadIdx.x] += sl[threadIdx.x + s];
      sp[threadIdx.x] += sp[threadIdx.x + s];
    }
    __syncthreads();
  }
  if (threadIdx.x == 0) { out[0] = sl[0] / (float)BB; out[1] = sp[0] / (float)BB; }
}

extern "C" void kernel_launch(void* const* d_in, const int* in_sizes, int n_in,
                              void* d_out, int out_size, void* d_ws, size_t ws_size,
                              hipStream_t stream) {
  const float* inputs  = (const float*)d_in[0];   // [B, D] fp32
  const int*   targets = (const int*)d_in[1];     // [B]    int32
  const float* centers = (const float*)d_in[2];   // [C, D] fp32
  float* out = (float*)d_out;                     // [2] (loss, prec)

  char* ws = (char*)d_ws;
  size_t o = 0;
  __bf16* in_hi = (__bf16*)(ws + o); o += (size_t)BB * DD * 2;
  __bf16* in_lo = (__bf16*)(ws + o); o += (size_t)BB * DD * 2;
  __bf16* ce_hi = (__bf16*)(ws + o); o += (size_t)CC * DD * 2;
  __bf16* ce_lo = (__bf16*)(ws + o); o += (size_t)CC * DD * 2;
  float* nin = (float*)(ws + o); o += (size_t)BB * 4;
  float* nc  = (float*)(ws + o); o += (size_t)CC * 4;
  int*   cnt = (int*)  (ws + o); o += (size_t)CC * 4;
  float* ap  = (float*)(ws + o); o += (size_t)BB * 4;
  float* an  = (float*)(ws + o); o += (size_t)BB * 4;

  k_split_bf16<<<(BB * DD + 255) / 256, 256, 0, stream>>>(inputs, in_hi, in_lo, BB * DD);
  k_split_bf16<<<(CC * DD + 255) / 256, 256, 0, stream>>>(centers, ce_hi, ce_lo, CC * DD);
  k_row_norm2<<<BB / 8, 256, 0, stream>>>(inputs, nin, BB);
  k_row_norm2<<<CC / 8, 256, 0, stream>>>(centers, nc, CC);
  k_zero_counts<<<1, 512, 0, stream>>>(cnt);
  k_count<<<BB / 256, 256, 0, stream>>>(targets, cnt);
  k_dist_minmax<<<BB / 64, 128, 0, stream>>>(in_hi, in_lo, ce_hi, ce_lo,
                                             nin, nc, cnt, targets, ap, an);
  k_finalize<<<1, 256, 0, stream>>>(ap, an, out);
}
